// Mask2FormerBase_16192026706163
// MI455X (gfx1250) — matrix operand order repaired and checked by reading, not compiled, not run on hardware
//
#include <hip/hip_runtime.h>

typedef __attribute__((ext_vector_type(16))) __bf16 v16bf;
typedef __attribute__((ext_vector_type(8)))  __bf16 v8bf;
typedef __attribute__((ext_vector_type(4)))  __bf16 v4bf;
typedef __attribute__((ext_vector_type(8)))  float  v8f;

#define LN_EPS  1e-5f
#define BN_EPS  1e-5f
#define RES_EPS 1e-6f

__device__ __forceinline__ v8f wmma_bf16(v16bf a, v16bf b, v8f c) {
  // D = A(16x32 bf16) * B(32x16 bf16) + C(16x16 f32)
  return __builtin_amdgcn_wmma_f32_16x16x32_bf16(false, a, false, b, (short)0, c, false, false);
}

// Concatenate two aligned 8xbf16 LDS chunks into a 16xbf16 fragment (2x ds_load_b128)
__device__ __forceinline__ v16bf ld_frag(const __bf16* p0, const __bf16* p1) {
  v8bf lo = *(const v8bf*)p0;
  v8bf hi = *(const v8bf*)p1;
  return __builtin_shufflevector(lo, hi, 0, 1, 2, 3, 4, 5, 6, 7, 8, 9, 10, 11, 12, 13, 14, 15);
}

// ---------------------------------------------------------------------------
// Backbone: direct conv 3x3 (pad 1) + eval BatchNorm + ReLU
// ---------------------------------------------------------------------------
__global__ void conv_bn_relu(const float* __restrict__ x, const float* __restrict__ w,
                             const float* __restrict__ cb,
                             const float* __restrict__ g, const float* __restrict__ bb,
                             const float* __restrict__ rm, const float* __restrict__ rv,
                             float* __restrict__ y,
                             int Bn, int Cin, int Hin, int Win,
                             int Cout, int Hout, int Wout, int stride)
{
  long idx = (long)blockIdx.x * blockDim.x + threadIdx.x;
  long total = (long)Bn * Cout * Hout * Wout;
  if (idx >= total) return;
  int wx = idx % Wout;
  int hy = (idx / Wout) % Hout;
  int co = (idx / ((long)Wout * Hout)) % Cout;
  int b  = idx / ((long)Wout * Hout * Cout);
  float acc = 0.f;
  for (int ci = 0; ci < Cin; ci++) {
    const float* xp = x + ((long)(b * Cin + ci) * Hin) * Win;
    const float* wp = w + ((long)(co * Cin + ci) * 3) * 3;
    for (int ky = 0; ky < 3; ky++) {
      int iy = hy * stride + ky - 1;
      if (iy < 0 || iy >= Hin) continue;
      for (int kx = 0; kx < 3; kx++) {
        int ix = wx * stride + kx - 1;
        if (ix < 0 || ix >= Win) continue;
        acc += xp[(long)iy * Win + ix] * wp[ky * 3 + kx];
      }
    }
  }
  acc += cb[co];
  acc = (acc - rm[co]) * g[co] * rsqrtf(rv[co] + BN_EPS) + bb[co];
  y[idx] = fmaxf(acc, 0.f);
}

// memory[b,hw,c] = features[b,c,hw] + pos[hw,c]
__global__ void mem_pos_kernel(const float* __restrict__ feat, const float* __restrict__ pos,
                               float* __restrict__ mem)
{
  long idx = (long)blockIdx.x * blockDim.x + threadIdx.x;
  if (idx >= 8L * 4096 * 256) return;
  int c  = idx % 256;
  int hw = (idx / 256) % 4096;
  int b  = idx / (256L * 4096);
  mem[idx] = feat[((long)b * 256 + c) * 4096 + hw] + pos[(long)hw * 256 + c];
}

__global__ void broadcast_queries(const float* __restrict__ q, float* __restrict__ out)
{
  int tok = blockIdx.x, c = threadIdx.x;           // 800 blocks x 256
  out[(long)tok * 256 + c] = q[(long)(tok % 100) * 256 + c];
}

// ---------------------------------------------------------------------------
// Block reduction helper (blockDim.x == 256)
// ---------------------------------------------------------------------------
__device__ __forceinline__ float block_reduce_sum(float v, float* sred)
{
  int c = threadIdx.x;
  sred[c] = v; __syncthreads();
  for (int s = 128; s > 0; s >>= 1) { if (c < s) sred[c] += sred[c + s]; __syncthreads(); }
  float r = sred[0]; __syncthreads();
  return r;
}

__global__ void ln_kernel(const float* __restrict__ x, const float* __restrict__ g,
                          const float* __restrict__ b, float* __restrict__ y)
{
  __shared__ float sred[256];
  int r = blockIdx.x, c = threadIdx.x;
  float v  = x[(long)r * 256 + c];
  float mu = block_reduce_sum(v, sred) * (1.f / 256.f);
  float d  = v - mu;
  float var = block_reduce_sum(d * d, sred) * (1.f / 256.f);
  y[(long)r * 256 + c] = d * rsqrtf(var + LN_EPS) * g[c] + b[c];
}

// Cross-layer softmax depth-attention residual over history (nd entries).
// nd==1 degenerates to identity copy (layer 0).
__global__ void depth_mix_kernel(const float* __restrict__ hist, const float* __restrict__ rq,
                                 const float* __restrict__ kscale, float* __restrict__ out, int nd)
{
  __shared__ float sred[256];
  __shared__ float dots[8];
  int tok = blockIdx.x, c = threadIdx.x;
  const long ds = 800L * 256;
  for (int d = 0; d < nd; d++) {
    float h   = hist[d * ds + (long)tok * 256 + c];
    float ss  = block_reduce_sum(h * h, sred);
    float rms = rsqrtf(ss * (1.f / 256.f) + RES_EPS);
    float dot = block_reduce_sum(rq[c] * h * rms * kscale[c], sred);
    if (c == 0) dots[d] = dot;
  }
  __syncthreads();
  float mx = -3e38f;
  for (int d = 0; d < nd; d++) mx = fmaxf(mx, dots[d]);
  float sum = 0.f, w[8];
  for (int d = 0; d < nd; d++) { w[d] = __expf(dots[d] - mx); sum += w[d]; }
  float acc = 0.f;
  for (int d = 0; d < nd; d++) acc += (w[d] / sum) * hist[d * ds + (long)tok * 256 + c];
  out[(long)tok * 256 + c] = acc;
}

// ---------------------------------------------------------------------------
// Generic tiled bf16-WMMA GEMM:  C = act(A @ op(B) + bias) (+= if accum)
//   A: [M,K] f32 row-major (lda), optionally batched (sA)
//   B: b_nk=1 -> W[N,K] row-major (ldb=K); b_nk=0 -> [K,N] row-major (ldb=N)
//   128x128 block tile, 8 waves (2x4), 64x32 per wave, K-step 32.
//   LDS rows padded to 40 bf16 (80 B, 16B multiple) so every lane fragment is
//   two aligned ds_load_b128's.
// ---------------------------------------------------------------------------
#define BM 128
#define BN 128
#define BK 32
#define KP 40   // padded LDS row (bf16 elements); 80 bytes

__global__ __launch_bounds__(256) void gemm_wmma(
    const float* __restrict__ A, long lda, long sA,
    const float* __restrict__ Bm, long ldb, long sB, int b_nk,
    const float* __restrict__ bias,
    float* __restrict__ C, long ldc, long sC,
    int M, int N, int K, int act, int accum)
{
  __shared__ __align__(16) __bf16 As[BM][KP];
  __shared__ __align__(16) __bf16 Bs[BN][KP];   // stored [n][k]

  int bz = blockIdx.z;
  A  += (long)bz * sA;
  Bm += (long)bz * sB;
  C  += (long)bz * sC;
  int m0 = blockIdx.y * BM, n0 = blockIdx.x * BN;
  int tid = threadIdx.x, lane = tid & 31, wave = tid >> 5;
  int half = lane >> 4, l16 = lane & 15;
  int wm = (wave >> 2) * 64, wn = (wave & 3) * 32;

  v8f acc[4][2];
#pragma unroll
  for (int mt = 0; mt < 4; mt++)
#pragma unroll
    for (int nt = 0; nt < 2; nt++)
      acc[mt][nt] = (v8f){0.f, 0.f, 0.f, 0.f, 0.f, 0.f, 0.f, 0.f};

  for (int k0 = 0; k0 < K; k0 += BK) {
    __syncthreads();
    // --- load A tile (128x32 f32 -> bf16 LDS, packed b64 stores) ---
    {
      int r  = tid >> 3;          // 0..31
      int kk = (tid & 7) * 4;     // 0..28
#pragma unroll
      for (int rr = 0; rr < BM; rr += 32) {
        int row = m0 + r + rr;
        float4 v = {0.f, 0.f, 0.f, 0.f};
        if (row < M && (k0 + kk) < K)
          v = *(const float4*)(A + (long)row * lda + k0 + kk);
        v4bf pk = { (__bf16)v.x, (__bf16)v.y, (__bf16)v.z, (__bf16)v.w };
        *(v4bf*)&As[r + rr][kk] = pk;
      }
    }
    // --- load B tile ---
    if (b_nk) {                   // W[N,K]: rows are output cols
      int r  = tid >> 3;
      int kk = (tid & 7) * 4;
#pragma unroll
      for (int rr = 0; rr < BN; rr += 32) {
        int nrow = n0 + r + rr;
        float4 v = {0.f, 0.f, 0.f, 0.f};
        if (nrow < N && (k0 + kk) < K)
          v = *(const float4*)(Bm + (long)nrow * ldb + k0 + kk);
        v4bf pk = { (__bf16)v.x, (__bf16)v.y, (__bf16)v.z, (__bf16)v.w };
        *(v4bf*)&Bs[r + rr][kk] = pk;
      }
    } else {                      // [K,N]: transpose while staging
      int kk = tid >> 5;          // 0..7
      int nn = (tid & 31) * 4;    // 0..124
#pragma unroll
      for (int kr = 0; kr < BK; kr += 8) {
        int kt = kk + kr;
        int nc = n0 + nn;
        float4 v = {0.f, 0.f, 0.f, 0.f};
        if ((k0 + kt) < K && nc < N)
          v = *(const float4*)(Bm + (long)(k0 + kt) * ldb + nc);
        Bs[nn + 0][kt] = (__bf16)v.x;
        Bs[nn + 1][kt] = (__bf16)v.y;
        Bs[nn + 2][kt] = (__bf16)v.z;
        Bs[nn + 3][kt] = (__bf16)v.w;
      }
    }
    __syncthreads();

    // --- fragments + WMMA ---
    // B fragments: k(e) = e + half*16  -> one contiguous 16-elt run
    v16bf bf0 = ld_frag(&Bs[wn + l16][half * 16],      &Bs[wn + l16][half * 16 + 8]);
    v16bf bf1 = ld_frag(&Bs[wn + 16 + l16][half * 16], &Bs[wn + 16 + l16][half * 16 + 8]);
#pragma unroll
    for (int mt = 0; mt < 4; mt++) {
      // A fragment: k(e) = e + half*16 -> one contiguous 16-elt run
      int row = wm + mt * 16 + l16;
      v16bf af = ld_frag(&As[row][half * 16], &As[row][half * 16 + 8]);
      acc[mt][0] = wmma_bf16(af, bf0, acc[mt][0]);
      acc[mt][1] = wmma_bf16(af, bf1, acc[mt][1]);
    }
  }

  // --- epilogue: bias / activation / residual-accumulate ---
#pragma unroll
  for (int mt = 0; mt < 4; mt++)
#pragma unroll
    for (int nt = 0; nt < 2; nt++)
#pragma unroll
      for (int i = 0; i < 8; i++) {
        int row = m0 + wm + mt * 16 + i + half * 8;
        int col = n0 + wn + nt * 16 + l16;
        if (row < M && col < N) {
          float v = acc[mt][nt][i];
          if (bias) v += bias[col];
          if (act == 1)      v = fmaxf(v, 0.f);
          else if (act == 2) v = 0.5f * v * (1.f + erff(v * 0.70710678118f)); // exact gelu
          long o = (long)row * ldc + col;
          C[o] = accum ? (C[o] + v) : v;
        }
      }
}

// ---------------------------------------------------------------------------
// Fused multi-head attention (flash-style online softmax), dh = 32, nh = 8.
// Block = (head, batch); 7 waves, one 16-query tile each (qlen = 100).
// V is staged transposed so P@V B-fragments are contiguous b128 loads.
// ---------------------------------------------------------------------------
__global__ __launch_bounds__(224) void attn_kernel(
    const float* __restrict__ qp, long q_rs, long q_bs,
    const float* __restrict__ kp, const float* __restrict__ vp, long kv_rs, long kv_bs,
    float* __restrict__ op, long o_rs, long o_bs,
    int qlen, int klen)
{
  __shared__ __align__(16) __bf16 Ks[32][40];      // [key][feat]
  __shared__ __align__(16) __bf16 Vst[32][40];     // [dh][key] (transposed)
  __shared__ __align__(16) __bf16 Ps[7][16][40];   // per-wave probs [qrow][key]

  int h = blockIdx.x, b = blockIdx.y;
  int tid = threadIdx.x, lane = tid & 31, wave = tid >> 5;
  int half = lane >> 4, l16 = lane & 15;
  int q0 = wave * 16;
  const float scale = 0.1767766952966369f;  // 1/sqrt(32)

  // Q fragment (A layout), scaled
  v16bf aq;
#pragma unroll
  for (int e = 0; e < 16; e++) {
    int kk  = e + half * 16;
    int row = q0 + l16;
    float v = (row < qlen) ? qp[(long)b * q_bs + (long)row * q_rs + h * 32 + kk] * scale : 0.f;
    aq[e] = (__bf16)v;
  }

  v8f o0 = (v8f){0.f,0.f,0.f,0.f,0.f,0.f,0.f,0.f};
  v8f o1 = o0;
  float mi[8], li[8];
#pragma unroll
  for (int i = 0; i < 8; i++) { mi[i] = -3e38f; li[i] = 0.f; }

  for (int c0 = 0; c0 < klen; c0 += 32) {
    __syncthreads();
    for (int idx = tid; idx < 1024; idx += 224) {
      int ky = idx >> 5, f = idx & 31;
      int gk = c0 + ky;
      float kv = 0.f, vv = 0.f;
      if (gk < klen) {
        long base = (long)b * kv_bs + (long)gk * kv_rs + h * 32 + f;
        kv = kp[base]; vv = vp[base];
      }
      Ks[ky][f]  = (__bf16)kv;
      Vst[f][ky] = (__bf16)vv;
    }
    __syncthreads();

    // logits: two 16x16 tiles over this 32-key chunk
    v16bf bk0 = ld_frag(&Ks[l16][half * 16],      &Ks[l16][half * 16 + 8]);
    v16bf bk1 = ld_frag(&Ks[16 + l16][half * 16], &Ks[16 + l16][half * 16 + 8]);
    v8f z = (v8f){0.f,0.f,0.f,0.f,0.f,0.f,0.f,0.f};
    v8f s0 = wmma_bf16(aq, bk0, z);
    v8f s1 = wmma_bf16(aq, bk1, z);

    // mask padded key columns
    if (c0 + l16 >= klen)      { for (int i = 0; i < 8; i++) s0[i] = -3e38f; }
    if (c0 + 16 + l16 >= klen) { for (int i = 0; i < 8; i++) s1[i] = -3e38f; }

    // online softmax update (row-wise across the 16 lanes of each half)
#pragma unroll
    for (int i = 0; i < 8; i++) {
      float cm = fmaxf(s0[i], s1[i]);
      cm = fmaxf(cm, __shfl_xor(cm, 1, 16));
      cm = fmaxf(cm, __shfl_xor(cm, 2, 16));
      cm = fmaxf(cm, __shfl_xor(cm, 4, 16));
      cm = fmaxf(cm, __shfl_xor(cm, 8, 16));
      float nm = fmaxf(mi[i], cm);
      float sc = __expf(mi[i] - nm);
      float p0 = __expf(s0[i] - nm);
      float p1 = __expf(s1[i] - nm);
      float rs = p0 + p1;
      rs += __shfl_xor(rs, 1, 16);
      rs += __shfl_xor(rs, 2, 16);
      rs += __shfl_xor(rs, 4, 16);
      rs += __shfl_xor(rs, 8, 16);
      li[i] = li[i] * sc + rs;
      o0[i] *= sc; o1[i] *= sc;
      mi[i] = nm;
      int rloc = i + half * 8;
      Ps[wave][rloc][l16]      = (__bf16)p0;
      Ps[wave][rloc][16 + l16] = (__bf16)p1;
    }

    // P fragment (wave-local LDS relayout C->A), V fragments, accumulate P@V
    v16bf pa  = ld_frag(&Ps[wave][l16][half * 16], &Ps[wave][l16][half * 16 + 8]);
    v16bf bv0 = ld_frag(&Vst[l16][half * 16],      &Vst[l16][half * 16 + 8]);
    v16bf bv1 = ld_frag(&Vst[16 + l16][half * 16], &Vst[16 + l16][half * 16 + 8]);
    o0 = wmma_bf16(pa, bv0, o0);
    o1 = wmma_bf16(pa, bv1, o1);
  }

#pragma unroll
  for (int i = 0; i < 8; i++) {
    int row = q0 + i + half * 8;
    if (row < qlen) {
      float inv = 1.f / li[i];
      long base = (long)b * o_bs + (long)row * o_rs + h * 32;
      op[base + l16]      = o0[i] * inv;
      op[base + 16 + l16] = o1[i] * inv;
    }
  }
}

// ---------------------------------------------------------------------------
// Host orchestration
// ---------------------------------------------------------------------------
static inline void launch_gemm(hipStream_t s,
    const float* A, long lda, long sA,
    const float* B, long ldb, long sB, int b_nk,
    const float* bias,
    float* C, long ldc, long sC,
    int M, int N, int K, int act, int accum, int batch)
{
  dim3 grid((N + BN - 1) / BN, (M + BM - 1) / BM, batch);
  gemm_wmma<<<grid, 256, 0, s>>>(A, lda, sA, B, ldb, sB, b_nk, bias, C, ldc, sC, M, N, K, act, accum);
}

extern "C" void kernel_launch(void* const* d_in, const int* in_sizes, int n_in,
                              void* d_out, int out_size, void* d_ws, size_t ws_size,
                              hipStream_t stream) {
  (void)in_sizes; (void)n_in; (void)out_size; (void)ws_size;
  const float* images  = (const float*)d_in[0];
  const float* c1_w = (const float*)d_in[1],  *c1_b = (const float*)d_in[2];
  const float* c1_g = (const float*)d_in[3],  *c1_bb = (const float*)d_in[4];
  const float* c1_m = (const float*)d_in[5],  *c1_v = (const float*)d_in[6];
  const float* c2_w = (const float*)d_in[7],  *c2_b = (const float*)d_in[8];
  const float* c2_g = (const float*)d_in[9],  *c2_bb = (const float*)d_in[10];
  const float* c2_m = (const float*)d_in[11], *c2_v = (const float*)d_in[12];
  const float* c3_w = (const float*)d_in[13], *c3_b = (const float*)d_in[14];
  const float* c3_g = (const float*)d_in[15], *c3_bb = (const float*)d_in[16];
  const float* c3_m = (const float*)d_in[17], *c3_v = (const float*)d_in[18];
  const float* queries = (const float*)d_in[19];
  const float* pos     = (const float*)d_in[20];
  const float* res_q   = (const float*)d_in[21];
  const float* kscale  = (const float*)d_in[22];
  const float* n1_g = (const float*)d_in[23], *n1_b = (const float*)d_in[24];
  const float* n2_g = (const float*)d_in[25], *n2_b = (const float*)d_in[26];
  const float* n3_g = (const float*)d_in[27], *n3_b = (const float*)d_in[28];
  const float* sa_w = (const float*)d_in[29], *sa_b = (const float*)d_in[30];
  const float* sa_ow = (const float*)d_in[31], *sa_ob = (const float*)d_in[32];
  const float* ca_w = (const float*)d_in[33], *ca_b = (const float*)d_in[34];
  const float* ca_ow = (const float*)d_in[35], *ca_ob = (const float*)d_in[36];
  const float* ff1_w = (const float*)d_in[37], *ff1_b = (const float*)d_in[38];
  const float* ff2_w = (const float*)d_in[39], *ff2_b = (const float*)d_in[40];
  const float* mask_w1 = (const float*)d_in[41], *mask_b1 = (const float*)d_in[42];
  const float* mask_w2 = (const float*)d_in[43], *mask_b2 = (const float*)d_in[44];
  const float* cls_w1 = (const float*)d_in[45], *cls_b1 = (const float*)d_in[46];
  const float* cls_w2 = (const float*)d_in[47], *cls_b2 = (const float*)d_in[48];
  float* out = (float*)d_out;

  // workspace layout (floats)
  float* ws = (float*)d_ws;
  float* x1       = ws;                    // 8*32*128*128 = 4194304
  float* x2       = x1 + 4194304;          // 8*64*64*64   = 2097152
  float* feat     = x2 + 2097152;          // 8*256*64*64  = 8388608
  float* mem      = feat + 8388608;        // 8*4096*256   = 8388608
  float* kbuf     = mem + 8388608;         // 8388608
  float* vbuf     = kbuf + 8388608;        // 8388608
  float* hist     = vbuf + 8388608;        // 7 * 800*256  = 1433600
  float* tn       = hist + 1433600;        // 204800
  float* qkv      = tn + 204800;           // 800*768 = 614400
  float* attn_out = qkv + 614400;          // 204800
  float* ffn_h    = attn_out + 204800;     // 800*2048 = 1638400
  float* membed   = ffn_h + 1638400;       // 204800
  float* tmp      = membed + 204800;       // 204800

  // ---- backbone ----
  {
    long n = 8L * 32 * 128 * 128;
    conv_bn_relu<<<(n + 255) / 256, 256, 0, stream>>>(images, c1_w, c1_b, c1_g, c1_bb, c1_m, c1_v,
                                                      x1, 8, 3, 256, 256, 32, 128, 128, 2);
    n = 8L * 64 * 64 * 64;
    conv_bn_relu<<<(n + 255) / 256, 256, 0, stream>>>(x1, c2_w, c2_b, c2_g, c2_bb, c2_m, c2_v,
                                                      x2, 8, 32, 128, 128, 64, 64, 64, 2);
    n = 8L * 256 * 64 * 64;
    conv_bn_relu<<<(n + 255) / 256, 256, 0, stream>>>(x2, c3_w, c3_b, c3_g, c3_bb, c3_m, c3_v,
                                                      feat, 8, 64, 64, 64, 256, 64, 64, 1);
    long nm = 8L * 4096 * 256;
    mem_pos_kernel<<<(nm + 255) / 256, 256, 0, stream>>>(feat, pos, mem);
  }
  broadcast_queries<<<800, 256, 0, stream>>>(queries, hist);

  const long HS = 800L * 256;
  // ---- decoder layers ----
  for (int i = 0; i < 6; i++) {
    float* t = hist + (long)(i + 1) * HS;
    depth_mix_kernel<<<800, 256, 0, stream>>>(hist, res_q + i * 256, kscale, t, i + 1);

    // self-attention
    ln_kernel<<<800, 256, 0, stream>>>(t, n1_g + i * 256, n1_b + i * 256, tn);
    launch_gemm(stream, tn, 256, 0, sa_w + (long)i * 196608, 256, 0, 1,
                sa_b + i * 768, qkv, 768, 0, 800, 768, 256, 0, 0, 1);
    attn_kernel<<<dim3(8, 8), 224, 0, stream>>>(qkv, 768, 76800,
                                                qkv + 256, qkv + 512, 768, 76800,
                                                attn_out, 256, 25600, 100, 100);
    launch_gemm(stream, attn_out, 256, 0, sa_ow + (long)i * 65536, 256, 0, 1,
                sa_ob + i * 256, t, 256, 0, 800, 256, 256, 0, 1, 1);

    // cross-attention (memory NOT normalized)
    ln_kernel<<<800, 256, 0, stream>>>(t, n2_g + i * 256, n2_b + i * 256, tn);
    launch_gemm(stream, tn, 256, 0, ca_w + (long)i * 196608, 256, 0, 1,
                ca_b + i * 768, qkv, 256, 0, 800, 256, 256, 0, 0, 1);
    launch_gemm(stream, mem, 256, 0, ca_w + (long)i * 196608 + 65536, 256, 0, 1,
                ca_b + i * 768 + 256, kbuf, 256, 0, 32768, 256, 256, 0, 0, 1);
    launch_gemm(stream, mem, 256, 0, ca_w + (long)i * 196608 + 131072, 256, 0, 1,
                ca_b + i * 768 + 512, vbuf, 256, 0, 32768, 256, 256, 0, 0, 1);
    attn_kernel<<<dim3(8, 8), 224, 0, stream>>>(qkv, 256, 25600,
                                                kbuf, vbuf, 256, 1048576,
                                                attn_out, 256, 25600, 100, 4096);
    launch_gemm(stream, attn_out, 256, 0, ca_ow + (long)i * 65536, 256, 0, 1,
                ca_ob + i * 256, t, 256, 0, 800, 256, 256, 0, 1, 1);

    // FFN (exact gelu)
    ln_kernel<<<800, 256, 0, stream>>>(t, n3_g + i * 256, n3_b + i * 256, tn);
    launch_gemm(stream, tn, 256, 0, ff1_w + (long)i * 524288, 256, 0, 1,
                ff1_b + i * 2048, ffn_h, 2048, 0, 800, 2048, 256, 2, 0, 1);
    launch_gemm(stream, ffn_h, 2048, 0, ff2_w + (long)i * 524288, 2048, 0, 1,
                ff2_b + i * 256, t, 256, 0, 800, 256, 2048, 0, 1, 1);
  }

  // ---- heads ----
  const float* last = hist + 6L * HS;
  launch_gemm(stream, last, 256, 0, mask_w1, 256, 0, 1, mask_b1, tmp, 256, 0,
              800, 256, 256, 1, 0, 1);
  launch_gemm(stream, tmp, 256, 0, mask_w2, 256, 0, 1, mask_b2, membed, 256, 0,
              800, 256, 256, 0, 0, 1);
  // pred_masks[b,q,hw] = membed[b,q,:] @ feat[b,:,hw]   (batched, B layout [K,N])
  launch_gemm(stream, membed, 256, 25600, feat, 4096, 1048576, 0, nullptr,
              out, 4096, 409600, 100, 4096, 256, 0, 0, 8);
  launch_gemm(stream, last, 256, 0, cls_w1, 256, 0, 1, cls_b1, tmp, 256, 0,
              800, 256, 256, 1, 0, 1);
  launch_gemm(stream, tmp, 256, 0, cls_w2, 256, 0, 1, cls_b2, out + 3276800, 80, 0,
              800, 80, 256, 0, 0, 1);
}
